// LSTM_49632642072633
// MI455X (gfx1250) — matrix operand order repaired and checked by reading, not compiled, not run on hardware
//
#include <hip/hip_runtime.h>
#include <hip/hip_bf16.h>
#include <stdint.h>

// ---------------------------------------------------------------------------
// 2-layer LSTM for MI455X (gfx1250, wave32, WMMA).
//   - GEMMs in f16 with f32 accumulation via v_wmma_f32_16x16x32_f16
//   - Tile staging via CDNA5 async copies (global_load_async_to_lds_b128),
//     double-buffered with counted s_wait_asynccnt to overlap copy & WMMA
//   - N=64 block tile (4 accumulators/wave): stays within VGPR budget,
//     no scratch spills (N=128 spilled accumulators in the K-loop)
//   - Recurrence: 32 persistent workgroups; each keeps a 256KB W_hh slice
//     resident in LDS (CDNA5: 320KB/WGP), c-state in registers, h exchanged
//     through a double-buffered L2-resident global buffer + grid barrier.
// ---------------------------------------------------------------------------

typedef __attribute__((ext_vector_type(16))) _Float16 v16h;
typedef __attribute__((ext_vector_type(8)))  _Float16 v8h;
typedef __attribute__((ext_vector_type(8)))  float    v8f;

#define NB  64      // batch
#define NC  512     // input channels
#define NT  512     // timesteps
#define NH  512     // hidden
#define NG  2048    // 4*NH (gates)

__device__ __forceinline__ float sigm(float x) { return 1.0f / (1.0f + __expf(-x)); }

// CDNA5 async copy: global -> LDS, 16 bytes, tracked by ASYNCcnt.
// VDST operand is the wave-relative LDS byte address (low 32 bits of the
// generic pointer per the flat-aperture truncation rule, ISA 10.2).
__device__ __forceinline__ void async_g2l_b128(unsigned lds_off, const void* gptr) {
  asm volatile("global_load_async_to_lds_b128 %0, %1, off"
               :: "v"(lds_off), "v"(gptr)
               : "memory");
}
__device__ __forceinline__ void wait_async_le3() {   // keep newest tile (3 ops) in flight
  asm volatile("s_wait_asynccnt 0x3" ::: "memory");
}
__device__ __forceinline__ void wait_async0() {
  asm volatile("s_wait_asynccnt 0x0" ::: "memory");
}
__device__ __forceinline__ unsigned lds_off(const void* p) {
  return (unsigned)(uintptr_t)p;
}

// ------------------------- weight conversion -------------------------------
__global__ void k_cvt_weights(const float* __restrict__ Wih1, const float* __restrict__ Whh1,
                              const float* __restrict__ bih1, const float* __restrict__ bhh1,
                              const float* __restrict__ Wih2, const float* __restrict__ Whh2,
                              const float* __restrict__ bih2, const float* __restrict__ bhh2,
                              _Float16* __restrict__ wih1h, _Float16* __restrict__ whh1h,
                              _Float16* __restrict__ wih2h, _Float16* __restrict__ whh2h,
                              float* __restrict__ bias1, float* __restrict__ bias2)
{
  const int n = NG * NH;
  for (int i = blockIdx.x * blockDim.x + threadIdx.x; i < n; i += gridDim.x * blockDim.x) {
    wih1h[i] = (_Float16)Wih1[i];
    whh1h[i] = (_Float16)Whh1[i];
    wih2h[i] = (_Float16)Wih2[i];
    whh2h[i] = (_Float16)Whh2[i];
  }
  for (int i = blockIdx.x * blockDim.x + threadIdx.x; i < NG; i += gridDim.x * blockDim.x) {
    bias1[i] = bih1[i] + bhh1[i];
    bias2[i] = bih2[i] + bhh2[i];
  }
}

// ------------------- x: (B,C,T) f32 -> xT: (B,T,C) f16 ---------------------
__global__ void k_transpose_cvt(const float* __restrict__ x, _Float16* __restrict__ xT)
{
  __shared__ float tile[32][33];
  const int b  = blockIdx.z;
  const int c0 = blockIdx.x * 32;
  const int t0 = blockIdx.y * 32;
  const int tx = threadIdx.x, ty = threadIdx.y;   // 32 x 8
  #pragma unroll
  for (int i = 0; i < 32; i += 8)
    tile[ty + i][tx] = x[(size_t)(b * NC + c0 + ty + i) * NT + t0 + tx];
  __syncthreads();
  #pragma unroll
  for (int i = 0; i < 32; i += 8)
    xT[(size_t)(b * NT + t0 + ty + i) * NC + c0 + tx] = (_Float16)tile[tx][ty + i];
}

// ------------- xp = A(f16)[M,512] @ W(f16)[2048,512]^T + bias --------------
// block = 256 threads (8 waves); tile M=128 (wave-per-16-rows), N=64, Kc=32.
// Double-buffered LDS staging with async copies: issue tile kc+1, then
// s_wait_asynccnt<=3 so only tile kc is waited for (3 async ops per wave
// per tile, in-order completion).
__global__ void k_proj_gemm(const _Float16* __restrict__ A,   // [M][512]
                            const _Float16* __restrict__ W,   // [2048][512] row-major
                            const float* __restrict__ bias,   // [2048]
                            float* __restrict__ Out)          // [M][2048]
{
  __shared__ __align__(32) _Float16 As[2][128][32];
  __shared__ __align__(32) _Float16 Bs[2][64][32];

  const int row0 = blockIdx.y * 128;
  const int col0 = blockIdx.x * 64;
  const int tid  = threadIdx.x;
  const int wave = tid >> 5;
  const int lane = tid & 31;
  const int half = lane >> 4;      // K-half select
  const int nlo  = lane & 15;

  // per-thread staging coordinates (fixed across K loop)
  const int ar  = tid >> 1;             // A row 0..127
  const int ac  = (tid & 1) * 16;       // A col 0 or 16
  const int brn = tid >> 2;             // B row 0..63
  const int bcn = (tid & 3) * 8;        // B col {0,8,16,24}
  const unsigned aoff[2] = { lds_off(&As[0][ar][ac]), lds_off(&As[1][ar][ac]) };
  const unsigned boff[2] = { lds_off(&Bs[0][brn][bcn]), lds_off(&Bs[1][brn][bcn]) };
  const _Float16* srcA = A + (size_t)(row0 + ar) * NH + ac;
  const _Float16* srcB = W + (size_t)(col0 + brn) * NH + bcn;

  // prologue: stage tile 0 into buffer 0
  async_g2l_b128(aoff[0],      srcA);
  async_g2l_b128(aoff[0] + 16, srcA + 8);
  async_g2l_b128(boff[0],      srcB);

  v8f acc[4] = {};

  for (int kc = 0; kc < 16; ++kc) {
    const int p = kc & 1;
    if (kc + 1 < 16) {
      // stage tile kc+1 into the other buffer, then wait only for tile kc
      const int q = p ^ 1;
      const _Float16* nA = srcA + (kc + 1) * 32;
      const _Float16* nB = srcB + (kc + 1) * 32;
      async_g2l_b128(aoff[q],      nA);
      async_g2l_b128(aoff[q] + 16, nA + 8);
      async_g2l_b128(boff[q],      nB);
      wait_async_le3();
    } else {
      wait_async0();
    }
    __syncthreads();

    // A fragment (16x32): lane m = lane%16,
    // elems 0..15 -> K = klo+0..15 contiguous, klo = 16*half
    v16h af;
    {
      const int m   = (wave << 4) + nlo;
      const int klo = half * 16;
      const v8h lo = *(const v8h*)&As[p][m][klo];
      const v8h hi = *(const v8h*)&As[p][m][klo + 8];
      #pragma unroll
      for (int e = 0; e < 8; ++e) { af[e] = lo[e]; af[8 + e] = hi[e]; }
    }
    #pragma unroll
    for (int nt = 0; nt < 4; ++nt) {
      // B fragment (32x16): column = lane%16, K = 16*half + 0..15 contiguous
      const int n  = nt * 16 + nlo;
      const int kb = half * 16;
      v16h bf = *(const v16h*)&Bs[p][n][kb];
      acc[nt] = __builtin_amdgcn_wmma_f32_16x16x32_f16(
          false, af, false, bf, (short)0, acc[nt], false, false);
    }
    __syncthreads();   // protect buffer p before it is re-staged at kc+2
  }

  // C/D layout: row = v + 8*half, col = lane%16
  #pragma unroll
  for (int nt = 0; nt < 4; ++nt) {
    const int col = col0 + nt * 16 + nlo;
    const float bv = bias[col];
    #pragma unroll
    for (int v = 0; v < 8; ++v) {
      const int row = row0 + (wave << 4) + v + half * 8;
      Out[(size_t)row * NG + col] = acc[nt][v] + bv;
    }
  }
}

// ----------------------------- recurrence ----------------------------------
// grid = 32 blocks: 4 batch-groups (16 rows) x 8 column-groups (64 h-cols).
// block = 128 threads (4 waves); wave w owns h-columns [cg*64+w*16, +16).
// W_hh slice (4 gates x 64 cols x 512 K, f16) = 256KB resident in LDS.
__global__ void k_lstm_rec(const float* __restrict__ xp,       // [B*T][2048]
                           const _Float16* __restrict__ Whh,   // [2048][512] f16
                           _Float16* __restrict__ hbuf,        // [2][B][H] f16 (buf0 zeroed)
                           _Float16* __restrict__ h_out,       // [B][T][H] f16 or null
                           float* __restrict__ y_out,          // [B][H][T] f32 or null
                           unsigned* __restrict__ sync_cnt)    // zeroed
{
  extern __shared__ __align__(32) _Float16 Wl[];   // [256][512]

  const int gb   = blockIdx.x;
  const int bg   = gb >> 3;           // batch group
  const int cg   = gb & 7;            // column group
  const int rb   = bg * 16;           // batch rows rb..rb+15
  const int tid  = threadIdx.x;
  const int wave = tid >> 5;          // 0..3
  const int lane = tid & 31;
  const int half = lane >> 4;
  const int nlo  = lane & 15;
  const unsigned nblk = gridDim.x;

  // preload W_hh slice via async copies:
  // LDS row r = q*64 + c  <->  global row q*512 + cg*64 + c
  for (int idx = tid; idx < 256 * 32; idx += 128) {
    const int r = idx >> 5;
    const int s = (idx & 31) << 4;
    const int q = r >> 6, c = r & 63;
    const _Float16* src = Whh + (size_t)(q * 512 + cg * 64 + c) * NH + s;
    async_g2l_b128(lds_off(&Wl[r * NH + s]), src);
    async_g2l_b128(lds_off(&Wl[r * NH + s + 8]), src + 8);
  }
  wait_async0();
  __syncthreads();

  float cst[8] = {0.f, 0.f, 0.f, 0.f, 0.f, 0.f, 0.f, 0.f};
  const int j = cg * 64 + wave * 16 + nlo;   // this lane's global h column

  for (int t = 0; t < NT; ++t) {
    const _Float16* hprev = hbuf + (size_t)(t & 1) * (NB * NH);
    _Float16*       hnext = hbuf + (size_t)((t + 1) & 1) * (NB * NH);

    v8f a0 = {}, a1 = {}, a2 = {}, a3 = {};
    const _Float16* arow = hprev + (size_t)(rb + nlo) * NH;   // A lane row = lane%16
    const _Float16* bcol = Wl + (size_t)(wave * 16 + nlo) * NH;

    #pragma unroll 4
    for (int kc = 0; kc < NH; kc += 32) {
      v16h af;
      {
        const int klo = kc + half * 16;
        const v8h lo = *(const v8h*)(arow + klo);
        const v8h hi = *(const v8h*)(arow + klo + 8);
        #pragma unroll
        for (int e = 0; e < 8; ++e) { af[e] = lo[e]; af[8 + e] = hi[e]; }
      }
      const int kb = kc + half * 16;
      v16h b0 = *(const v16h*)(bcol + 0 * 64 * NH + kb);
      v16h b1 = *(const v16h*)(bcol + 1 * 64 * NH + kb);
      v16h b2 = *(const v16h*)(bcol + 2 * 64 * NH + kb);
      v16h b3 = *(const v16h*)(bcol + 3 * 64 * NH + kb);
      a0 = __builtin_amdgcn_wmma_f32_16x16x32_f16(false, af, false, b0, (short)0, a0, false, false);
      a1 = __builtin_amdgcn_wmma_f32_16x16x32_f16(false, af, false, b1, (short)0, a1, false, false);
      a2 = __builtin_amdgcn_wmma_f32_16x16x32_f16(false, af, false, b2, (short)0, a2, false, false);
      a3 = __builtin_amdgcn_wmma_f32_16x16x32_f16(false, af, false, b3, (short)0, a3, false, false);
    }

    // pointwise LSTM cell: all 4 gates for column j live in this lane
    #pragma unroll
    for (int v = 0; v < 8; ++v) {
      const int r = rb + v + half * 8;                          // batch row (C/D layout)
      const float* xr = xp + ((size_t)r * NT + t) * NG;
      const float gi = a0[v] + xr[0 * NH + j];
      const float gf = a1[v] + xr[1 * NH + j];
      const float gg = a2[v] + xr[2 * NH + j];
      const float go = a3[v] + xr[3 * NH + j];
      const float iv = sigm(gi);
      const float fv = sigm(gf);
      const float gv = tanhf(gg);
      const float ov = sigm(go);
      const float cv = fv * cst[v] + iv * gv;
      cst[v] = cv;
      const float hv = ov * tanhf(cv);
      hnext[(size_t)r * NH + j] = (_Float16)hv;
      if (h_out) h_out[((size_t)r * NT + t) * NH + j] = (_Float16)hv;
      if (y_out) y_out[((size_t)r * NH + j) * NT + t] = hv;
    }

    // grid barrier: release h writes, arrive, spin until all blocks arrived
    __threadfence();
    __syncthreads();
    if (tid == 0) {
      __hip_atomic_fetch_add(sync_cnt, 1u, __ATOMIC_ACQ_REL, __HIP_MEMORY_SCOPE_AGENT);
      const unsigned target = nblk * (unsigned)(t + 1);
      while (__hip_atomic_load(sync_cnt, __ATOMIC_ACQUIRE, __HIP_MEMORY_SCOPE_AGENT) < target) {
        __builtin_amdgcn_s_sleep(1);
      }
    }
    __syncthreads();
  }
}

// ------------------------------- host side ---------------------------------
extern "C" void kernel_launch(void* const* d_in, const int* in_sizes, int n_in,
                              void* d_out, int out_size, void* d_ws, size_t ws_size,
                              hipStream_t stream)
{
  (void)in_sizes; (void)n_in; (void)out_size; (void)ws_size;
  const float* x    = (const float*)d_in[0];
  const float* Wih1 = (const float*)d_in[1];
  const float* Whh1 = (const float*)d_in[2];
  const float* bih1 = (const float*)d_in[3];
  const float* bhh1 = (const float*)d_in[4];
  const float* Wih2 = (const float*)d_in[5];
  const float* Whh2 = (const float*)d_in[6];
  const float* bih2 = (const float*)d_in[7];
  const float* bhh2 = (const float*)d_in[8];
  float* out = (float*)d_out;

  // workspace carve-up (all offsets 256B aligned)
  uint8_t* ws = (uint8_t*)d_ws;
  size_t o = 0;
  auto carve = [&](size_t bytes) { void* p = ws + o; o += (bytes + 255) & ~(size_t)255; return p; };
  _Float16* wih1h = (_Float16*)carve((size_t)NG * NH * 2);
  _Float16* whh1h = (_Float16*)carve((size_t)NG * NH * 2);
  _Float16* wih2h = (_Float16*)carve((size_t)NG * NH * 2);
  _Float16* whh2h = (_Float16*)carve((size_t)NG * NH * 2);
  float*    bias1 = (float*)carve((size_t)NG * 4);
  float*    bias2 = (float*)carve((size_t)NG * 4);
  _Float16* xT    = (_Float16*)carve((size_t)NB * NT * NC * 2);
  _Float16* h1    = (_Float16*)carve((size_t)NB * NT * NH * 2);
  _Float16* hbuf  = (_Float16*)carve((size_t)2 * NB * NH * 2);
  unsigned* sync  = (unsigned*)carve(256);
  float*    xp    = (float*)carve((size_t)NB * NT * NG * 4);   // 268 MB

  const size_t ldsRec = (size_t)256 * NH * sizeof(_Float16);   // 256 KB
  (void)hipFuncSetAttribute((const void*)k_lstm_rec,
                            hipFuncAttributeMaxDynamicSharedMemorySize, (int)ldsRec);

  // 1) convert weights / fold biases
  k_cvt_weights<<<1024, 256, 0, stream>>>(Wih1, Whh1, bih1, bhh1, Wih2, Whh2, bih2, bhh2,
                                          wih1h, whh1h, wih2h, whh2h, bias1, bias2);
  // 2) x (B,C,T) -> xT (B,T,C) f16
  k_transpose_cvt<<<dim3(NC / 32, NT / 32, NB), dim3(32, 8), 0, stream>>>(x, xT);

  // ---- layer 1 ----
  k_proj_gemm<<<dim3(NG / 64, (NB * NT) / 128), 256, 0, stream>>>(xT, wih1h, bias1, xp);
  hipMemsetAsync(hbuf, 0, (size_t)2 * NB * NH * 2, stream);
  hipMemsetAsync(sync, 0, sizeof(unsigned), stream);
  k_lstm_rec<<<32, 128, ldsRec, stream>>>(xp, whh1h, hbuf, h1, nullptr, sync);

  // ---- layer 2 ----
  k_proj_gemm<<<dim3(NG / 64, (NB * NT) / 128), 256, 0, stream>>>(h1, wih2h, bias2, xp);
  hipMemsetAsync(hbuf, 0, (size_t)2 * NB * NH * 2, stream);
  hipMemsetAsync(sync, 0, sizeof(unsigned), stream);
  k_lstm_rec<<<32, 128, ldsRec, stream>>>(xp, whh2h, hbuf, nullptr, out, sync);
}
